// MeshSimplifier_3100966388296
// MI455X (gfx1250) — matrix operand order repaired and checked by reading, not compiled, not run on hardware
//
#include <hip/hip_runtime.h>
#include <hip/hip_bf16.h>

// ---------------- problem constants ----------------
#define N_NODES 12000
#define N_EDGES 384000
#define HDIM    64
#define M_KEEP  6000          // (1-RATIO)*N
#define K_TOP   15
#define P_PAIRS 105           // C(15,2)
#define MP      (M_KEEP * P_PAIRS)   // 630000
#define SORT_S  16384

// ---------------- CDNA5 WMMA types ----------------
typedef __attribute__((ext_vector_type(16))) __bf16 v16bf;
typedef __attribute__((ext_vector_type(8)))  float  v8f;

// async global->LDS path (gfx1250): guarded so absence of the builtin
// falls back to a plain LDS copy instead of breaking the build.
#if defined(__has_builtin)
#  if __has_builtin(__builtin_amdgcn_global_load_async_to_lds_b128) && \
      __has_builtin(__builtin_amdgcn_s_wait_asynccnt)
#    define HAVE_ASYNC_LDS 1
#  endif
#endif
#ifndef HAVE_ASYNC_LDS
#  define HAVE_ASYNC_LDS 0
#endif

// builtin signature (from hipcc diagnostic): (v4i AS1*, v4i AS3*, imm int, imm int)
typedef __attribute__((__vector_size__(16))) int v4i;
typedef __attribute__((address_space(1))) v4i g_v4i;
typedef __attribute__((address_space(3))) v4i l_v4i;

__device__ __forceinline__ unsigned short f2bf(float f) {  // used only in one-shot repack
  unsigned u = __float_as_uint(f);
  u += 0x7FFFu + ((u >> 16) & 1u);
  return (unsigned short)(u >> 16);
}

__device__ __forceinline__ v8f wmma_bf16(v16bf a, v16bf b, v8f c) {
  // D(16x16 f32) = A(16x32 bf16) * B(32x16 bf16) + C
  return __builtin_amdgcn_wmma_f32_16x16x32_bf16(false, a, false, b, (short)0, c,
                                                 false, false);
}

// Build a 16x32 bf16 A-fragment from an f32 row-major matrix (ld floats).
// CDNA5 layout: lanes0-15 M=lane K=[k0..k0+7]+[k0+16..k0+23];
//               lanes16-31 M=lane-16 K=[k0+8..k0+15]+[k0+24..k0+31].
// Native __bf16 casts -> v_cvt_pk_bf16_f32 (2 elem/op) instead of bit-twiddle VALU.
__device__ __forceinline__ v16bf load_a_f32(const float* S, int ld, int m0, int k0,
                                            int lane) {
  int m  = m0 + (lane & 15);
  int kb = k0 + ((lane & 16) >> 1);          // +8 for upper half-wave
  const float* sp = S + m * ld + kb;
  v16bf a;
#pragma unroll
  for (int t = 0; t < 8; ++t) {
    a[t]     = (__bf16)sp[t];
    a[t + 8] = (__bf16)sp[t + 16];
  }
  return a;
}

// B fragments are pre-packed: frag(kt,nt) -> [lane][16 bf16] contiguous.
__device__ __forceinline__ v16bf load_b_pack(const unsigned short* P, int kt, int nt,
                                             int lane) {
  return *(const v16bf*)(P + ((kt * 4 + nt) * 32 + lane) * 16);
}

__device__ __forceinline__ unsigned ordkey(float f) {
  unsigned u = __float_as_uint(f);
  return u ^ ((u & 0x80000000u) ? 0xFFFFFFFFu : 0x80000000u);
}
__device__ __forceinline__ float fromord(unsigned u) {
  u = (u & 0x80000000u) ? (u ^ 0x80000000u) : ~u;
  return __uint_as_float(u);
}
__device__ __forceinline__ float sigm(float x) { return 1.f / (1.f + expf(-x)); }

// ---------------- weight repack: f32 64x64 -> bf16 WMMA-B fragments ----------------
// per lane: elements e=0..15 map K = kt*32 + 8*(lane>=16) + e + (e&8), N = nt*16 + lane%16.
__global__ void pack_weight(const float* __restrict__ W, unsigned short* __restrict__ P) {
  int f = threadIdx.x >> 5, lane = threadIdx.x & 31;   // 8 fragments x 32 lanes
  int kt = f >> 2, nt = f & 3;
  int n  = nt * 16 + (lane & 15);
  int kb = kt * 32 + ((lane & 16) >> 1);
  unsigned short* dst = P + ((long)f * 32 + lane) * 16;
#pragma unroll
  for (int e = 0; e < 16; ++e) dst[e] = f2bf(W[(kb + e + (e & 8)) * HDIM + n]);
}

// ---------------- stage 1: node embed ----------------
__global__ void node_embed(const float* __restrict__ pos, const float* __restrict__ w_in,
                           float* __restrict__ h) {
  int t = blockIdx.x * blockDim.x + threadIdx.x;
  if (t >= N_NODES * HDIM) return;
  int i = t >> 6, j = t & 63;
  float a = pos[i * 3 + 0] * w_in[0 * HDIM + j] +
            pos[i * 3 + 1] * w_in[1 * HDIM + j] +
            pos[i * 3 + 2] * w_in[2 * HDIM + j];
  h[t] = fmaxf(a, 0.f);
}

// ---------------- stage 2: edge segment-sum scatter ----------------
__global__ void scatter_edges(const int* __restrict__ ei, const float* __restrict__ h,
                              float* __restrict__ agg) {
  int t = blockIdx.x * blockDim.x + threadIdx.x;
  if (t >= N_EDGES * HDIM) return;
  int e = t >> 6, j = t & 63;
  int s = ei[e], d = ei[N_EDGES + e];
  atomicAdd(&agg[d * HDIM + j], h[s * HDIM + j]);
}

// ---------------- stage 3: GNN layer, WMMA: relu(h@Ws + agg@Wn) ----------------
// B fragments (16KB for the two weight packs) are staged into LDS once per block
// via the gfx1250 async global->LDS path, then served to all 8 waves from ds_load.
__global__ void gnn_layer(const float* __restrict__ h, const float* __restrict__ agg,
                          const unsigned short* __restrict__ Wsp,
                          const unsigned short* __restrict__ Wnp,
                          float* __restrict__ hout) {
  __shared__ unsigned short sW[8192];        // [0..4095]=Ws pack, [4096..8191]=Wn pack
  int tid  = threadIdx.x;
  int lane = tid & 31;
  int wv   = (blockIdx.x * blockDim.x + tid) >> 5;
  int m0   = wv * 16;

#if HAVE_ASYNC_LDS
  {
    g_v4i* g0 = (g_v4i*)(void*)Wsp;
    g_v4i* g1 = (g_v4i*)(void*)Wnp;
    l_v4i* l0 = (l_v4i*)(void*)sW;
    for (int t = tid; t < 512; t += 256) {   // 512 x b128 per 8KB pack
      __builtin_amdgcn_global_load_async_to_lds_b128(g0 + t, l0 + t, 0, 0);
      __builtin_amdgcn_global_load_async_to_lds_b128(g1 + t, l0 + 512 + t, 0, 0);
    }
    __builtin_amdgcn_s_wait_asynccnt(0);
  }
#else
  {
    uint4* d = (uint4*)sW;
    const uint4* s0 = (const uint4*)Wsp;
    const uint4* s1 = (const uint4*)Wnp;
    for (int t = tid; t < 512; t += 256) { d[t] = s0[t]; d[512 + t] = s1[t]; }
  }
#endif
  __syncthreads();
  if (m0 >= N_NODES) return;                 // wave-uniform exit (EXEC stays full)

  const unsigned short* ws = sW;
  const unsigned short* wn = sW + 4096;
  v16bf ah0 = load_a_f32(h,   HDIM, m0, 0,  lane);
  v16bf ah1 = load_a_f32(h,   HDIM, m0, 32, lane);
  v16bf ag0 = load_a_f32(agg, HDIM, m0, 0,  lane);
  v16bf ag1 = load_a_f32(agg, HDIM, m0, 32, lane);
#pragma unroll
  for (int nt = 0; nt < 4; ++nt) {
    v8f c = {};
    c = wmma_bf16(ah0, load_b_pack(ws, 0, nt, lane), c);
    c = wmma_bf16(ah1, load_b_pack(ws, 1, nt, lane), c);
    c = wmma_bf16(ag0, load_b_pack(wn, 0, nt, lane), c);
    c = wmma_bf16(ag1, load_b_pack(wn, 1, nt, lane), c);
    int col   = nt * 16 + (lane & 15);
    int rbase = m0 + ((lane & 16) >> 1);     // +8 for upper half-wave
#pragma unroll
    for (int r = 0; r < 8; ++r) hout[(rbase + r) * HDIM + col] = fmaxf(c[r], 0.f);
  }
}

// ---------------- stage 4: node sample probs ----------------
__global__ void node_probs(const float* __restrict__ h, const float* __restrict__ w_out,
                           float* __restrict__ probs) {
  int i = blockIdx.x * blockDim.x + threadIdx.x;
  if (i >= N_NODES) return;
  float a = 0.f;
#pragma unroll 8
  for (int j = 0; j < HDIM; ++j) a += h[i * HDIM + j] * w_out[j];
  probs[i] = sigm(a);
}

// ---------------- stage 5: exact top-M sort (single WGP, 128KB LDS bitonic) ----------
__global__ void topk_sort(const float* __restrict__ probs, const float* __restrict__ pos,
                          float* __restrict__ out_idx, float* __restrict__ out_sp,
                          int* __restrict__ mask, int* __restrict__ newid,
                          int* __restrict__ sidx) {
  extern __shared__ unsigned long long key[];
  int tid = threadIdx.x;
  for (int i = tid; i < SORT_S; i += blockDim.x) {
    if (i < N_NODES) {
      unsigned u = ordkey(probs[i]);
      key[i] = (((unsigned long long)(~u)) << 32) | (unsigned)i;  // desc prob, asc idx
    } else key[i] = ~0ull;
  }
  __syncthreads();
  for (int k = 2; k <= SORT_S; k <<= 1) {
    for (int j = k >> 1; j > 0; j >>= 1) {
      for (int i = tid; i < SORT_S; i += blockDim.x) {
        int ixj = i ^ j;
        if (ixj > i) {
          bool up = (i & k) == 0;
          unsigned long long a = key[i], b = key[ixj];
          if ((a > b) == up) { key[i] = b; key[ixj] = a; }
        }
      }
      __syncthreads();
    }
  }
  for (int r = tid; r < M_KEEP; r += blockDim.x) {
    int idx = (int)(key[r] & 0xFFFFFFFFu);
    out_idx[r] = (float)idx;
    sidx[r] = idx; mask[idx] = 1; newid[idx] = r;
    out_sp[r * 3 + 0] = pos[idx * 3 + 0];
    out_sp[r * 3 + 1] = pos[idx * 3 + 1];
    out_sp[r * 3 + 2] = pos[idx * 3 + 2];
  }
}

// ---------------- stage 6: edge MLP + dense adj scatter (adj lives in 192MB L2) -----
__global__ void edge_kernel(const int* __restrict__ ei, const int* __restrict__ mask,
                            const int* __restrict__ newid, const float* __restrict__ sp,
                            const float* __restrict__ w1, const float* __restrict__ b1,
                            const float* __restrict__ w2,
                            float* __restrict__ o_src, float* __restrict__ o_dst,
                            float* __restrict__ o_prob, float* __restrict__ adj) {
  int e = blockIdx.x * blockDim.x + threadIdx.x;
  if (e >= N_EDGES) return;
  int s = ei[e], d = ei[N_EDGES + e];
  bool valid = mask[s] && mask[d];
  int sn = valid ? newid[s] : M_KEEP;
  int dn = valid ? newid[d] : M_KEEP;
  o_src[e] = (float)sn; o_dst[e] = (float)dn;
  float f[6];
  if (sn < M_KEEP) { f[0]=sp[sn*3]; f[1]=sp[sn*3+1]; f[2]=sp[sn*3+2]; }
  else             { f[0]=f[1]=f[2]=0.f; }
  if (dn < M_KEEP) { f[3]=sp[dn*3]; f[4]=sp[dn*3+1]; f[5]=sp[dn*3+2]; }
  else             { f[3]=f[4]=f[5]=0.f; }
  float acc = 0.f;
#pragma unroll 4
  for (int c = 0; c < HDIM; ++c) {
    float a = b1[c];
#pragma unroll
    for (int q = 0; q < 6; ++q) a += f[q] * w1[q * HDIM + c];
    acc += fmaxf(a, 0.f) * w2[c];
  }
  float p = valid ? sigm(acc) : 0.f;
  o_prob[e] = p;
  adj[(size_t)sn * (M_KEEP + 1) + dn] = p;   // .at[].set semantics: last writer wins
}

// ---------------- stage 7: per-row top-15 of adj (wave per row) ----------------
__global__ void row_topk(const float* __restrict__ adj, float* __restrict__ tv,
                         int* __restrict__ ti) {
  __shared__ float sv[32 * K_TOP];
  __shared__ int   si[32 * K_TOP];
  int row = blockIdx.x, lane = threadIdx.x;
  const float* r = adj + (size_t)row * (M_KEEP + 1);
  float lv[K_TOP]; int li[K_TOP];
#pragma unroll
  for (int s = 0; s < K_TOP; ++s) { lv[s] = -1.f; li[s] = 0x7FFFFFFF; }
  for (int col = lane; col < M_KEEP; col += 32) {
    float v = r[col];
    if (v > lv[K_TOP - 1] || (v == lv[K_TOP - 1] && col < li[K_TOP - 1])) {
      int s = K_TOP - 1;
      while (s > 0 && (v > lv[s - 1] || (v == lv[s - 1] && col < li[s - 1]))) {
        lv[s] = lv[s - 1]; li[s] = li[s - 1]; --s;
      }
      lv[s] = v; li[s] = col;
    }
  }
#pragma unroll
  for (int s = 0; s < K_TOP; ++s) { sv[lane * K_TOP + s] = lv[s]; si[lane * K_TOP + s] = li[s]; }
  __syncthreads();
  if (lane == 0) {
    for (int s = 0; s < K_TOP; ++s) {
      float bv = -1e30f; int bi = 0x7FFFFFFF, best = -1;
      for (int q = 0; q < 32 * K_TOP; ++q) {
        int ix = si[q];
        if (ix == 0x7FFFFFFF) continue;
        float v = sv[q];
        if (v > bv || (v == bv && ix < bi)) { bv = v; bi = ix; best = q; }
      }
      si[best] = 0x7FFFFFFF;
      tv[row * K_TOP + s] = bv; ti[row * K_TOP + s] = bi;
    }
  }
}

// ---------------- stage 8: face MLP (fused, WMMA for the 64x64 layer) ---------------
__global__ void face_kernel(const float* __restrict__ tv, const int* __restrict__ ti,
                            const float* __restrict__ adj, const float* __restrict__ sp,
                            const float* __restrict__ w1, const float* __restrict__ b1,
                            const unsigned short* __restrict__ w2p,
                            const float* __restrict__ b2, const float* __restrict__ w3,
                            float* __restrict__ o_faces, float* __restrict__ o_cprob,
                            float* __restrict__ o_fprob, int* __restrict__ valid_ws,
                            int* __restrict__ vcnt) {
  __shared__ float sX[16 * 12];
  __shared__ float sF1[16 * 72];
  __shared__ float sF2[16 * 72];
  __shared__ float sCP[16];
  int lane = threadIdx.x;
  int c0 = blockIdx.x * 16;
  __builtin_prefetch(w2p, 0, 0);
  if (lane < 16) {
    int c = c0 + lane;
    int i = c / P_PAIRS, p = c - i * P_PAIRS;
    int j = 0, rem = p;                       // decode triu pair (j<k) from p
    while (rem >= 14 - j) { rem -= 14 - j; ++j; }
    int k = j + 1 + rem;
    int n1 = ti[i * K_TOP + j], n2 = ti[i * K_TOP + k];
    float a12 = adj[(size_t)n1 * (M_KEEP + 1) + n2];
    int valid = a12 > 0.f;
    float cp = valid ? tv[i * K_TOP + j] * tv[i * K_TOP + k] * a12 * (1.f / 3.f) : 0.f;
    o_faces[(size_t)c * 3 + 0] = (float)i;
    o_faces[(size_t)c * 3 + 1] = (float)n1;
    o_faces[(size_t)c * 3 + 2] = (float)n2;
    o_cprob[c] = cp;
    sCP[lane] = cp;
    valid_ws[c] = valid;
    if (valid) atomicAdd(vcnt, 1);
    sX[lane * 12 + 0] = sp[i  * 3 + 0]; sX[lane * 12 + 1] = sp[i  * 3 + 1];
    sX[lane * 12 + 2] = sp[i  * 3 + 2]; sX[lane * 12 + 3] = sp[n1 * 3 + 0];
    sX[lane * 12 + 4] = sp[n1 * 3 + 1]; sX[lane * 12 + 5] = sp[n1 * 3 + 2];
    sX[lane * 12 + 6] = sp[n2 * 3 + 0]; sX[lane * 12 + 7] = sp[n2 * 3 + 1];
    sX[lane * 12 + 8] = sp[n2 * 3 + 2];
  }
  __syncthreads();
  {   // layer 1: x(16x9) @ w1(9x64), scalar (K=9 too small for the matrix pipe)
    int r = lane & 15, cb = (lane & 16) << 1;   // cols 0..31 / 32..63
    for (int cc = 0; cc < 32; ++cc) {
      int col = cb + cc;
      float a = b1[col];
#pragma unroll
      for (int q = 0; q < 9; ++q) a += sX[r * 12 + q] * w1[q * HDIM + col];
      sF1[r * 72 + col] = fmaxf(a, 0.f);
    }
  }
  __syncthreads();
  // layer 2: fh(16x64) @ w2(64x64) on WMMA bf16
  v16bf a0 = load_a_f32(sF1, 72, 0, 0,  lane);
  v16bf a1 = load_a_f32(sF1, 72, 0, 32, lane);
#pragma unroll
  for (int nt = 0; nt < 4; ++nt) {
    v8f c = {};
    c = wmma_bf16(a0, load_b_pack(w2p, 0, nt, lane), c);
    c = wmma_bf16(a1, load_b_pack(w2p, 1, nt, lane), c);
    int col = nt * 16 + (lane & 15);
    float bb = b2[col];
    int rb = (lane & 16) >> 1;
#pragma unroll
    for (int r = 0; r < 8; ++r) sF2[(rb + r) * 72 + col] = fmaxf(c[r] + bb, 0.f);
  }
  __syncthreads();
  if (lane < 16) {
    float acc = 0.f;
#pragma unroll 8
    for (int q = 0; q < HDIM; ++q) acc += sF2[lane * 72 + q] * w3[q];
    o_fprob[c0 + lane] = sigm(acc) * sCP[lane];
  }
}

// ---------------- stage 9: quantile threshold via radix-select ----------------
// sel[]: 0=V 1=lo 2=hi 3=frac 4=prefix0 5=rank0 6=prefix1 7=rank1 8=keyLo 9=keyHi
__global__ void init_select(const int* __restrict__ vcnt, unsigned* __restrict__ sel) {
  int V = *vcnt;
  float p = 0.5f * (float)max(V - 1, 0);
  int lo = min(max((int)floorf(p), 0), MP - 1);
  int hi = min(max((int)ceilf(p),  0), MP - 1);
  sel[0] = (unsigned)V; sel[1] = (unsigned)lo; sel[2] = (unsigned)hi;
  sel[3] = __float_as_uint(p - (float)lo);
  sel[4] = 0u; sel[5] = (unsigned)lo;
  sel[6] = 0u; sel[7] = (unsigned)hi;
}

__global__ void hist_pass(const float* __restrict__ fp, const int* __restrict__ valid,
                          const unsigned* __restrict__ sel, unsigned* __restrict__ hist,
                          int pass, int which) {
  int c = blockIdx.x * blockDim.x + threadIdx.x;
  if (c >= MP) return;
  unsigned key = valid[c] ? ordkey(fp[c]) : 0xFFFFFFFFu;   // invalid -> sorts last
  unsigned prefix = sel[4 + which * 2];
  int shift = 24 - pass * 8;
  bool match = (pass == 0) || ((key >> (shift + 8)) == prefix);
  if (match) atomicAdd(&hist[(key >> shift) & 0xFFu], 1u);
}

__global__ void pick_bin(unsigned* __restrict__ sel, const unsigned* __restrict__ hist,
                         int pass, int which) {
  unsigned prefix = sel[4 + which * 2], rank = sel[5 + which * 2];
  unsigned cum = 0, bin = 255;
  for (int b = 0; b < 256; ++b) {
    unsigned c = hist[b];
    if (cum + c > rank) { bin = (unsigned)b; rank -= cum; break; }
    cum += c;
  }
  prefix = (prefix << 8) | bin;
  sel[4 + which * 2] = prefix; sel[5 + which * 2] = rank;
  if (pass == 3) sel[8 + which] = prefix;     // full 32-bit ordered key of the rank-th value
}

__global__ void mask_kernel(const float* __restrict__ fp, const int* __restrict__ valid,
                            const unsigned* __restrict__ sel, float* __restrict__ o_mask) {
  int c = blockIdx.x * blockDim.x + threadIdx.x;
  if (c >= MP) return;
  float svlo = fromord(sel[8]), svhi = fromord(sel[9]);
  float frac = __uint_as_float(sel[3]);
  int V = (int)sel[0];
  float thr = (V > 0) ? (svlo * (1.f - frac) + svhi * frac) : __int_as_float(0x7F800000);
  o_mask[c] = (valid[c] && fp[c] > thr) ? 1.f : 0.f;
}

// ---------------- launch ----------------
extern "C" void kernel_launch(void* const* d_in, const int* in_sizes, int n_in,
                              void* d_out, int out_size, void* d_ws, size_t ws_size,
                              hipStream_t stream) {
  const float* pos    = (const float*)d_in[0];
  const int*   ei     = (const int*)  d_in[1];
  const float* w_in   = (const float*)d_in[2];
  const float* ps_ws  = (const float*)d_in[3];
  const float* ps_wn  = (const float*)d_in[4];
  const float* w_out  = (const float*)d_in[5];
  const float* ep_w1  = (const float*)d_in[6];
  const float* ep_b1  = (const float*)d_in[7];
  const float* ep_w2  = (const float*)d_in[8];
  const float* fc_w1  = (const float*)d_in[9];
  const float* fc_b1  = (const float*)d_in[10];
  const float* fc_w2  = (const float*)d_in[11];
  const float* fc_b2  = (const float*)d_in[12];
  const float* fc_w3  = (const float*)d_in[13];

  float* out = (float*)d_out;
  float* o_sidx  = out;                         // M
  float* o_probs = out + 6000;                  // N
  float* o_sp    = out + 18000;                 // M*3
  float* o_src   = out + 36000;                 // E
  float* o_dst   = out + 420000;                // E
  float* o_eprob = out + 804000;                // E
  float* o_faces = out + 1188000;               // MP*3
  float* o_cprob = out + 3078000;               // MP
  float* o_fprob = out + 3708000;               // MP
  float* o_mask  = out + 4338000;               // MP

  // workspace carve-up
  char* ws = (char*)d_ws;
  size_t off = 0;
  auto take = [&](size_t bytes) { size_t o = off; off = (off + bytes + 255) & ~(size_t)255; return o; };
  float*          h0    = (float*)(ws + take((size_t)N_NODES * HDIM * 4));
  float*          h1    = (float*)(ws + take((size_t)N_NODES * HDIM * 4));
  float*          agg   = (float*)(ws + take((size_t)N_NODES * HDIM * 4));
  unsigned short* wpack = (unsigned short*)(ws + take(7 * 4096 * 2));   // 7 matrices
  int*            nmask = (int*)(ws + take((size_t)N_NODES * 4));
  int*            newid = (int*)(ws + take((size_t)N_NODES * 4));
  int*            sidx  = (int*)(ws + take((size_t)M_KEEP * 4));
  float*          tvv   = (float*)(ws + take((size_t)M_KEEP * K_TOP * 4));
  int*            tii   = (int*)(ws + take((size_t)M_KEEP * K_TOP * 4));
  int*            vflag = (int*)(ws + take((size_t)MP * 4));
  int*            vcnt  = (int*)(ws + take(64));
  unsigned*       sel   = (unsigned*)(ws + take(64 * 4));
  unsigned*       hist  = (unsigned*)(ws + take(256 * 4));
  float*          adj   = (float*)(ws + take((size_t)(M_KEEP + 1) * (M_KEEP + 1) * 4));
  (void)off; (void)ws_size; (void)in_sizes; (void)n_in; (void)out_size;

  // repack 64x64 weights into bf16 WMMA-B fragment order (0..2: ps_ws, 3..5: ps_wn, 6: fc_w2)
  for (int l = 0; l < 3; ++l) {
    pack_weight<<<1, 256, 0, stream>>>(ps_ws + (size_t)l * HDIM * HDIM, wpack + (size_t)l * 4096);
    pack_weight<<<1, 256, 0, stream>>>(ps_wn + (size_t)l * HDIM * HDIM, wpack + (size_t)(3 + l) * 4096);
  }
  pack_weight<<<1, 256, 0, stream>>>(fc_w2, wpack + (size_t)6 * 4096);

  // GNN
  node_embed<<<(N_NODES * HDIM + 255) / 256, 256, 0, stream>>>(pos, w_in, h0);
  const float* hc = h0; float* hn = h1;
  int gnnBlocks = ((N_NODES / 16) + 7) / 8;    // 8 waves/block, 16 rows/wave
  for (int l = 0; l < 3; ++l) {
    (void)hipMemsetAsync(agg, 0, (size_t)N_NODES * HDIM * 4, stream);
    scatter_edges<<<(N_EDGES * HDIM) / 256, 256, 0, stream>>>(ei, hc, agg);
    gnn_layer<<<gnnBlocks, 256, 0, stream>>>(hc, agg,
                                             wpack + (size_t)l * 4096,
                                             wpack + (size_t)(3 + l) * 4096,
                                             hn);
    const float* t = hc; hc = hn; hn = (float*)t;
  }

  // sample
  node_probs<<<(N_NODES + 255) / 256, 256, 0, stream>>>(hc, w_out, o_probs);
  (void)hipMemsetAsync(nmask, 0, (size_t)N_NODES * 4, stream);
  topk_sort<<<1, 1024, SORT_S * 8, stream>>>(o_probs, pos, o_sidx, o_sp, nmask, newid, sidx);

  // edges -> dense adj (144MB, resident in 192MB L2)
  (void)hipMemsetAsync(adj, 0, (size_t)(M_KEEP + 1) * (M_KEEP + 1) * 4, stream);
  (void)hipMemsetAsync(vcnt, 0, 4, stream);
  edge_kernel<<<(N_EDGES + 255) / 256, 256, 0, stream>>>(ei, nmask, newid, o_sp,
                                                         ep_w1, ep_b1, ep_w2,
                                                         o_src, o_dst, o_eprob, adj);

  // per-row top-15
  row_topk<<<M_KEEP, 32, 0, stream>>>(adj, tvv, tii);

  // fused face MLP (WMMA)
  face_kernel<<<MP / 16, 32, 0, stream>>>(tvv, tii, adj, o_sp,
                                          fc_w1, fc_b1, wpack + (size_t)6 * 4096,
                                          fc_b2, fc_w3,
                                          o_faces, o_cprob, o_fprob, vflag, vcnt);

  // quantile threshold: two radix-selects (lo, hi) then mask
  init_select<<<1, 1, 0, stream>>>(vcnt, sel);
  for (int which = 0; which < 2; ++which) {
    for (int pass = 0; pass < 4; ++pass) {
      (void)hipMemsetAsync(hist, 0, 256 * 4, stream);
      hist_pass<<<(MP + 255) / 256, 256, 0, stream>>>(o_fprob, vflag, sel, hist, pass, which);
      pick_bin<<<1, 1, 0, stream>>>(sel, hist, pass, which);
    }
  }
  mask_kernel<<<(MP + 255) / 256, 256, 0, stream>>>(o_fprob, vflag, sel, o_mask);
}
